// GGATGRU_74741020885534
// MI455X (gfx1250) — compile-verified, matrix operand order checked by reading, not run on hardware
//
#include <hip/hip_runtime.h>
#include <hip/hip_bf16.h>
#include <stdint.h>
#include <stddef.h>

// ---------------------------------------------------------------------------
// Types for CDNA5 WMMA (wave32, 16x16x32 bf16 -> f32 accumulate)
// ---------------------------------------------------------------------------
typedef __attribute__((ext_vector_type(16))) __bf16        v16bf;
typedef __attribute__((ext_vector_type(8)))  float         v8f;
typedef __attribute__((ext_vector_type(8)))  unsigned int  v8u;

#define NEG_SLOPE 0.2f

// ---------------------------------------------------------------------------
// Float atomic max via CAS (segment-max for softmax)
// ---------------------------------------------------------------------------
__device__ inline void atomicMaxF(float* addr, float val) {
  unsigned int* ua = (unsigned int*)addr;
  unsigned int old = *ua;
  while (__uint_as_float(old) < val) {
    unsigned int assumed = old;
    old = atomicCAS(ua, assumed, __float_as_uint(val));
    if (old == assumed) break;
  }
}

// ---------------------------------------------------------------------------
// WMMA GEMM:  C[N,M] = act( concat(A1[K1], A2[K2]) @ W[K,M] + bias )
//   - 128 threads = 4 wave32 waves; each wave owns a 16-row strip.
//   - MT column tiles of 16 (M = MT*16), all accumulated in registers.
//   - K chunked by 32; A tile (64x32) and W^T chunk (Mx32) staged in LDS
//     as bf16 with padded stride 36 to avoid bank conflicts.
//   - act: 0 = none, 1 = sigmoid, 2 = tanh
// ---------------------------------------------------------------------------
template<int MT>
__global__ __launch_bounds__(128)
void k_gemm(const float* __restrict__ A1, int K1,
            const float* __restrict__ A2, int K2,
            const float* __restrict__ W,  const float* __restrict__ bias,
            float* __restrict__ C, int Nrows, int act)
{
  constexpr int M = MT * 16;
  const int K = K1 + K2;

  __shared__ __bf16 lds_a[64 * 36];
  __shared__ __bf16 lds_b[M * 36];

  const int tid  = threadIdx.x;
  const int w    = tid >> 5;        // wave id 0..3
  const int lane = tid & 31;
  const int l16  = lane & 15;
  const int half = lane >> 4;       // 0 or 1 (selects K sub-range)
  const int rowblk = blockIdx.x * 64;

  v8f acc[MT];
  const v8f vzero = {0.f,0.f,0.f,0.f,0.f,0.f,0.f,0.f};
#pragma unroll
  for (int t = 0; t < MT; ++t) acc[t] = vzero;

  for (int kc = 0; kc < K; kc += 32) {
    __syncthreads();
    // ---- stage A tile: 64 rows x 32 k, fp32 -> bf16 --------------------
    {
      const int r  = tid >> 1;            // 0..63
      const int kb = (tid & 1) * 16;      // 0 or 16
      const int grow = rowblk + r;
#pragma unroll
      for (int i = 0; i < 16; ++i) {
        const int k = kc + kb + i;
        float v = 0.f;
        if (grow < Nrows) {
          v = (k < K1) ? A1[(size_t)grow * K1 + k]
                       : A2[(size_t)grow * K2 + (k - K1)];
        }
        lds_a[r * 36 + kb + i] = (__bf16)v;
      }
    }
    // ---- stage W^T chunk: lds_b[m][k] = W[kc+k][m] ----------------------
    for (int i = tid; i < 32 * M; i += 128) {
      const int k = i / M;
      const int m = i % M;                 // coalesced over m
      lds_b[m * 36 + k] = (__bf16)W[(size_t)(kc + k) * M + m];
    }
    __syncthreads();

    // ---- A fragment for this wave's 16-row strip ------------------------
    v16bf afrag;
    {
      v8u au;
#pragma unroll
      for (int j = 0; j < 8; ++j) {
        const int kb = (j < 4 ? 2 * j : 16 + 2 * (j - 4)) + half * 8;
        au[j] = *(const unsigned int*)&lds_a[(w * 16 + l16) * 36 + kb];
      }
      afrag = __builtin_bit_cast(v16bf, au);
    }
    // ---- MT WMMA ops, one per 16-col tile -------------------------------
#pragma unroll
    for (int t = 0; t < MT; ++t) {
      v8u bu;
#pragma unroll
      for (int j = 0; j < 8; ++j) {
        const int kb = (j < 4 ? 2 * j : 16 + 2 * (j - 4)) + half * 8;
        bu[j] = *(const unsigned int*)&lds_b[(t * 16 + l16) * 36 + kb];
      }
      v16bf bfrag = __builtin_bit_cast(v16bf, bu);
      acc[t] = __builtin_amdgcn_wmma_f32_16x16x32_bf16(
          false, afrag, false, bfrag, (short)0, acc[t], false, false);
    }
  }

  // ---- epilogue: bias + activation + store ------------------------------
  const int strip = rowblk + w * 16;
#pragma unroll
  for (int t = 0; t < MT; ++t) {
    const int col = t * 16 + l16;
    const float bv = bias ? bias[col] : 0.f;
#pragma unroll
    for (int v = 0; v < 8; ++v) {
      const int row = strip + v + half * 8;
      if (row < Nrows) {
        float x = acc[t][v] + bv;
        if (act == 1)      x = 1.f / (1.f + __expf(-x));
        else if (act == 2) x = tanhf(x);
        C[(size_t)row * M + col] = x;
      }
    }
  }
}

// ---------------------------------------------------------------------------
// Elementwise / GAT helper kernels
// ---------------------------------------------------------------------------
__global__ void k_fill(float* __restrict__ p, long long n, float v) {
  long long i = (long long)blockIdx.x * blockDim.x + threadIdx.x;
  if (i < n) p[i] = v;
}

// alpha_src / alpha_dst per (node, head)
__global__ void k_alpha(const float* __restrict__ xp,
                        const float* __restrict__ as,
                        const float* __restrict__ ad,
                        float* __restrict__ aS, float* __restrict__ aD,
                        int N, int H, int Cc)
{
  long long id = (long long)blockIdx.x * blockDim.x + threadIdx.x;
  if (id >= (long long)N * H) return;
  const int n = (int)(id / H), h = (int)(id % H);
  const float* xr = xp + (size_t)n * H * Cc + (size_t)h * Cc;
  float s = 0.f, d = 0.f;
  for (int c = 0; c < Cc; ++c) {
    const float xv = xr[c];
    s += xv * as[h * Cc + c];
    d += xv * ad[h * Cc + c];
  }
  aS[id] = s; aD[id] = d;
}

__device__ inline void edge_sd(const long long* ei, long long E, long long e,
                               int& s, int& d) {
  if (e < E) { s = (int)ei[e]; d = (int)ei[E + e]; }
  else       { s = d = (int)(e - E); }             // self loop
}

__global__ void k_edge_max(const long long* __restrict__ ei, long long E, int N,
                           const float* __restrict__ aS,
                           const float* __restrict__ aD,
                           float* __restrict__ emax, int H)
{
  long long e = (long long)blockIdx.x * blockDim.x + threadIdx.x;
  if (e >= E + N) return;
  int s, d; edge_sd(ei, E, e, s, d);
  for (int h = 0; h < H; ++h) {
    float v = aS[(size_t)s * H + h] + aD[(size_t)d * H + h];
    v = v > 0.f ? v : NEG_SLOPE * v;
    atomicMaxF(&emax[(size_t)d * H + h], v);
  }
}

__global__ void k_edge_exp(const long long* __restrict__ ei, long long E, int N,
                           const float* __restrict__ aS,
                           const float* __restrict__ aD,
                           const float* __restrict__ emax,
                           float* __restrict__ expb,
                           float* __restrict__ denom, int H)
{
  long long e = (long long)blockIdx.x * blockDim.x + threadIdx.x;
  if (e >= E + N) return;
  int s, d; edge_sd(ei, E, e, s, d);
  for (int h = 0; h < H; ++h) {
    float v = aS[(size_t)s * H + h] + aD[(size_t)d * H + h];
    v = v > 0.f ? v : NEG_SLOPE * v;
    const float wgt = __expf(v - emax[(size_t)d * H + h]);
    expb[(size_t)e * H + h] = wgt;
    atomicAdd(&denom[(size_t)d * H + h], wgt);
  }
}

// one thread per (edge, channel): scatter xp[src]*alpha into out[dst]
__global__ void k_edge_aggr(const long long* __restrict__ ei, long long E, int N,
                            const float* __restrict__ xp,
                            const float* __restrict__ expb,
                            const float* __restrict__ denom,
                            float* __restrict__ out, int H, int Cc)
{
  const int HC = H * Cc;
  long long id = (long long)blockIdx.x * blockDim.x + threadIdx.x;
  const long long tot = (E + N) * (long long)HC;
  if (id >= tot) return;
  const long long e = id / HC;
  const int c = (int)(id % HC);
  const int h = c / Cc;
  int s, d; edge_sd(ei, E, e, s, d);
  const float alpha = expb[(size_t)e * H + h] /
                      (denom[(size_t)d * H + h] + 1e-16f);
  atomicAdd(&out[(size_t)d * HC + c], xp[(size_t)s * HC + c] * alpha);
}

__global__ void k_bias_act(float* __restrict__ buf, const float* __restrict__ bias,
                           long long n, int M, int do_elu)
{
  long long i = (long long)blockIdx.x * blockDim.x + threadIdx.x;
  if (i >= n) return;
  float v = buf[i] + bias[(int)(i % M)];
  if (do_elu) v = v > 0.f ? v : (__expf(v) - 1.f);
  buf[i] = v;
}

__global__ void k_mul(const float* __restrict__ a, const float* __restrict__ b,
                      float* __restrict__ c, long long n)
{
  long long i = (long long)blockIdx.x * blockDim.x + threadIdx.x;
  if (i < n) c[i] = a[i] * b[i];
}

// out = (1-z)*h_in + z*h_hat
__global__ void k_gru_final(const float* __restrict__ z,
                            const float* __restrict__ hin,
                            const float* __restrict__ hhat,
                            float* __restrict__ out, long long n)
{
  long long i = (long long)blockIdx.x * blockDim.x + threadIdx.x;
  if (i < n) {
    const float zz = z[i];
    out[i] = (1.f - zz) * hin[i] + zz * hhat[i];
  }
}

// ---------------------------------------------------------------------------
// Host orchestration
// ---------------------------------------------------------------------------
extern "C" void kernel_launch(void* const* d_in, const int* in_sizes, int n_in,
                              void* d_out, int out_size, void* d_ws, size_t ws_size,
                              hipStream_t stream)
{
  (void)in_sizes; (void)n_in; (void)out_size; (void)ws_size;

  const int       N = 50000;
  const long long E = 800000;

  const float*     x    = (const float*)d_in[0];
  const long long* ei   = (const long long*)d_in[1];
  const float *W_tr=(const float*)d_in[2],  *b_tr=(const float*)d_in[3];
  const float *W1  =(const float*)d_in[4],  *as1 =(const float*)d_in[5];
  const float *ad1 =(const float*)d_in[6],  *b1  =(const float*)d_in[7];
  const float *Wr1 =(const float*)d_in[8],  *br1 =(const float*)d_in[9];
  const float *Wz1 =(const float*)d_in[10], *bz1 =(const float*)d_in[11];
  const float *Wh1 =(const float*)d_in[12], *bh1 =(const float*)d_in[13];
  const float *Wp2 =(const float*)d_in[14], *bp2 =(const float*)d_in[15];
  const float *W2  =(const float*)d_in[16], *as2 =(const float*)d_in[17];
  const float *ad2 =(const float*)d_in[18], *b2  =(const float*)d_in[19];
  const float *Wr2 =(const float*)d_in[20], *br2 =(const float*)d_in[21];
  const float *Wz2 =(const float*)d_in[22], *bz2 =(const float*)d_in[23];
  const float *Wh2 =(const float*)d_in[24], *bh2 =(const float*)d_in[25];
  const float *Wp3 =(const float*)d_in[26], *bp3 =(const float*)d_in[27];
  const float *W3  =(const float*)d_in[28], *as3 =(const float*)d_in[29];
  const float *ad3 =(const float*)d_in[30], *b3  =(const float*)d_in[31];
  const float *Wr3 =(const float*)d_in[32], *br3 =(const float*)d_in[33];
  const float *Wz3 =(const float*)d_in[34], *bz3 =(const float*)d_in[35];
  const float *Wh3 =(const float*)d_in[36], *bh3 =(const float*)d_in[37];

  // ---- workspace carve-up ------------------------------------------------
  char*  ws  = (char*)d_ws;
  size_t off = 0;
  auto  carve = [&](size_t bytes) -> float* {
    float* p = (float*)(ws + off);
    off += (bytes + 255) & ~(size_t)255;
    return p;
  };
  const size_t BIG = (size_t)N * 256 * sizeof(float);
  float* B_hin = carve(BIG);                  // h_in (x0 / projections)
  float* B_xp  = carve(BIG);                  // GAT xp / h_hat
  float* B_h   = carve(BIG);                  // GAT aggregation / h
  float* B_t0  = carve(BIG);                  // r
  float* B_t1  = carve(BIG);                  // z
  float* B_t2  = carve(BIG);                  // rhin / layer output x
  float* aS    = carve((size_t)N * 8 * sizeof(float));
  float* aD    = carve((size_t)N * 8 * sizeof(float));
  float* emax  = carve((size_t)N * 8 * sizeof(float));
  float* denom = carve((size_t)N * 8 * sizeof(float));
  float* expb  = carve((size_t)(E + N) * 8 * sizeof(float));

  // ---- launch helpers ----------------------------------------------------
  auto gemm = [&](const float* A1, int K1, const float* A2, int K2,
                  const float* W, const float* b, float* C, int M, int act) {
    dim3 g((N + 63) / 64), blk(128);
    if (M == 256)      hipLaunchKernelGGL((k_gemm<16>), g, blk, 0, stream, A1,K1,A2,K2,W,b,C,N,act);
    else if (M == 128) hipLaunchKernelGGL((k_gemm<8>),  g, blk, 0, stream, A1,K1,A2,K2,W,b,C,N,act);
    else               hipLaunchKernelGGL((k_gemm<4>),  g, blk, 0, stream, A1,K1,A2,K2,W,b,C,N,act);
  };
  auto fill = [&](float* p, long long n, float v) {
    hipLaunchKernelGGL(k_fill, dim3((unsigned)((n + 255) / 256)), dim3(256), 0, stream, p, n, v);
  };
  auto gat = [&](const float* xin, const float* W, const float* asv,
                 const float* adv, const float* bias, int H, int Cc,
                 float* xp, float* out, int do_elu) {
    const int HC = H * Cc;
    gemm(xin, H == 1 ? 128 : 256, nullptr, 0, W, nullptr, xp, HC, 0);
    // NOTE: K of this GEMM is the input dim; handled below by explicit call
  };
  (void)gat;

  auto edge_pipeline = [&](const float* xp, const float* asv, const float* adv,
                           const float* bias, int H, int Cc, float* out, int do_elu) {
    const int HC = H * Cc;
    const long long Et = E + N;
    hipLaunchKernelGGL(k_alpha, dim3((unsigned)(((long long)N * H + 255) / 256)),
                       dim3(256), 0, stream, xp, asv, adv, aS, aD, N, H, Cc);
    fill(emax, (long long)N * H, -3.0e38f);
    fill(denom, (long long)N * H, 0.f);
    fill(out, (long long)N * HC, 0.f);
    hipLaunchKernelGGL(k_edge_max, dim3((unsigned)((Et + 255) / 256)), dim3(256),
                       0, stream, ei, E, N, aS, aD, emax, H);
    hipLaunchKernelGGL(k_edge_exp, dim3((unsigned)((Et + 255) / 256)), dim3(256),
                       0, stream, ei, E, N, aS, aD, emax, expb, denom, H);
    const long long tot = Et * (long long)HC;
    hipLaunchKernelGGL(k_edge_aggr, dim3((unsigned)((tot + 255) / 256)), dim3(256),
                       0, stream, ei, E, N, xp, expb, denom, out, H, Cc);
    hipLaunchKernelGGL(k_bias_act, dim3((unsigned)(((long long)N * HC + 255) / 256)),
                       dim3(256), 0, stream, out, bias, (long long)N * HC, HC, do_elu);
  };

  auto gru = [&](const float* h, const float* hin, int D,
                 const float* Wr, const float* br, const float* Wz, const float* bz,
                 const float* Wh, const float* bh, float* out) {
    const long long n = (long long)N * D;
    gemm(h, D, hin, D, Wr, br, B_t0, D, 1);                  // r = sigmoid(...)
    gemm(h, D, hin, D, Wz, bz, B_t1, D, 1);                  // z = sigmoid(...)
    hipLaunchKernelGGL(k_mul, dim3((unsigned)((n + 255) / 256)), dim3(256),
                       0, stream, B_t0, hin, B_t2, n);       // rhin = r*h_in
    gemm(h, D, B_t2, D, Wh, bh, B_xp, D, 2);                 // h_hat = tanh(...)
    hipLaunchKernelGGL(k_gru_final, dim3((unsigned)((n + 255) / 256)), dim3(256),
                       0, stream, B_t1, hin, B_xp, out, n);
  };

  // ======================= Layer 1 =======================
  gemm(x, 128, nullptr, 0, W_tr, b_tr, B_hin, 256, 0);       // x0 = x@W_tr+b (h1)
  gemm(B_hin, 256, nullptr, 0, W1, nullptr, B_xp, 256, 0);   // xp1
  edge_pipeline(B_xp, as1, ad1, b1, 8, 32, B_h, 1);          // GAT1 + ELU -> B_h
  gru(B_h, B_hin, 256, Wr1, br1, Wz1, bz1, Wh1, bh1, B_t2);  // x1 -> B_t2

  // ======================= Layer 2 =======================
  gemm(B_t2, 256, nullptr, 0, W2, nullptr, B_xp, 128, 0);    // xp2
  edge_pipeline(B_xp, as2, ad2, b2, 4, 32, B_h, 1);          // GAT2 + ELU -> B_h
  gemm(B_t2, 256, nullptr, 0, Wp2, bp2, B_hin, 128, 0);      // h_in2 = h2@Wp2+b
  gru(B_h, B_hin, 128, Wr2, br2, Wz2, bz2, Wh2, bh2, B_t2);  // x2 -> B_t2

  // ======================= Layer 3 =======================
  gemm(B_t2, 128, nullptr, 0, W3, nullptr, B_xp, 64, 0);     // xp3
  edge_pipeline(B_xp, as3, ad3, b3, 1, 64, B_h, 0);          // GAT3 (no ELU)
  gemm(B_t2, 128, nullptr, 0, Wp3, bp3, B_hin, 64, 0);       // h_in3 = h3@Wp3+b
  gru(B_h, B_hin, 64, Wr3, br3, Wz3, bz3, Wh3, bh3, (float*)d_out);
}
